// GraphNodeEmbedding_44246753083821
// MI455X (gfx1250) — matrix operand order repaired and checked by reading, not compiled, not run on hardware
//
#include <hip/hip_runtime.h>
#include <math.h>

// ---------------------------------------------------------------------------
// Types for CDNA5 WMMA (wave32, V_WMMA_F32_16X16X32_BF16)
// ---------------------------------------------------------------------------
typedef __bf16 bf16_t;
typedef __attribute__((ext_vector_type(16))) __bf16 v16bf;
typedef __attribute__((ext_vector_type(8)))  float  v8f;

struct alignas(16) U4 { unsigned int a, b, c, d; };
union AFrag { v16bf v; U4 u[2]; };

__device__ __forceinline__ v16bf make_frag(const bf16_t* p0, const bf16_t* p1) {
  AFrag f;
  f.u[0] = *(const U4*)p0;   // 8 contiguous bf16 (16B)
  f.u[1] = *(const U4*)p1;   // 8 contiguous bf16 (16B)
  return f.v;
}

__device__ __forceinline__ v8f vzero8() {
  v8f z;
#pragma unroll
  for (int i = 0; i < 8; ++i) z[i] = 0.f;
  return z;
}

__device__ __forceinline__ float sigmoidf_(float x) {
  return 1.f / (1.f + __expf(-x));
}

// ---------------------------------------------------------------------------
// B-fragment packing into LDS (bf16).
// Fragment (kt, nt) covers K = 32*kt..+31, N = 16*nt..+15.
// ISA B layout (32x16, 16-bit): lane holds column N = lane&15,
// element j holds K = 32*kt + 8*(lane>>4) + (j&7) + 16*(j>>3).
// Packed: dst[((kt*NT + nt)*32 + lane)*16 + j]
// transposed=false: W is (K x 64) row-major, B[k][n] = W[k*ldw + n]
// transposed=true : W is (n-major), B[k][n] = W[n*ldw + k]   (for W.T GEMMs)
// ---------------------------------------------------------------------------
__device__ __forceinline__ void pack_b_frags(bf16_t* dst,
                                             const float* __restrict__ W,
                                             int nkt, int NT, bool transposed,
                                             int ldw, int tid, int nthreads) {
  const int total = nkt * NT * 512;
  for (int idx = tid; idx < total; idx += nthreads) {
    int j    = idx & 15;
    int lane = (idx >> 4) & 31;
    int frag = idx >> 9;
    int nt   = frag % NT;
    int kt   = frag / NT;
    int half = lane >> 4;
    int n = nt * 16 + (lane & 15);
    int k = kt * 32 + half * 8 + (j & 7) + (j >> 3) * 16;
    float w = transposed ? W[n * ldw + k] : W[k * ldw + n];
    dst[idx] = (bf16_t)w;
  }
}

__device__ __forceinline__ v16bf load_bfrag(const bf16_t* base, int lane) {
  const bf16_t* p = base + lane * 16;
  return make_frag(p, p + 8);
}

// A-run pointer for the concatenated edge feature vector
// ei = [ ns[src] (64) | ns[tgt] (64) | ef (32) ], all bf16.
// Runs are 8 contiguous elements starting at multiples of 8 -> never cross
// a segment boundary, always 16B aligned.
__device__ __forceinline__ const bf16_t* ei_ptr(const bf16_t* __restrict__ nsb,
                                                const bf16_t* __restrict__ efb,
                                                int s, int t, long e, int k) {
  if (k < 64)  return nsb + (long)s * 64 + k;
  if (k < 128) return nsb + (long)t * 64 + (k - 64);
  return efb + e * 32 + (k - 128);
}

// ---------------------------------------------------------------------------
// Edge kernel (persistent waves): per 16-edge tile
//   ei (16x160 bf16) -> h = relu(ei@W1+b1) -> out = h@W2+b2 (msg & att paths)
//   scatter draft*sigmoid(gate) into msum via f32 atomics.
// 56 v_wmma_f32_16x16x32_bf16 per tile.
// ---------------------------------------------------------------------------
#define EK_WAVES 8
#define HSTRIDE 72

__global__ void __launch_bounds__(256) edge_kernel(
    const bf16_t* __restrict__ nsb, const bf16_t* __restrict__ efb,
    const int* __restrict__ edges,
    const float* __restrict__ mw1, const float* __restrict__ mb1,
    const float* __restrict__ mw2, const float* __restrict__ mb2,
    const float* __restrict__ aw1, const float* __restrict__ ab1,
    const float* __restrict__ aw2, const float* __restrict__ ab2,
    float* __restrict__ msum, long numTiles)
{
  // Packed bf16 weight fragments: msg_w1[0:10240) att_w1[10240:20480)
  // msg_w2[20480:24576) att_w2[24576:28672)
  __shared__ __align__(16) bf16_t smw[28672];
  __shared__ __align__(16) bf16_t smh[EK_WAVES * 16 * HSTRIDE];

  const int tid = threadIdx.x;
  pack_b_frags(smw +     0, mw1, 5, 4, false, 64, tid, 256);
  pack_b_frags(smw + 10240, aw1, 5, 4, false, 64, tid, 256);
  pack_b_frags(smw + 20480, mw2, 2, 4, false, 64, tid, 256);
  pack_b_frags(smw + 24576, aw2, 2, 4, false, 64, tid, 256);
  __syncthreads();

  const int wave = tid >> 5;
  const int lane = tid & 31;
  const int half = lane >> 4;
  const int lm   = lane & 15;      // A layout: row; C layout: column
  bf16_t* hbase = smh + wave * (16 * HSTRIDE);

  const long stride = (long)gridDim.x * EK_WAVES;
  for (long tile = (long)blockIdx.x * EK_WAVES + wave; tile < numTiles;
       tile += stride) {
    long e = tile * 16 + lm;       // this lane's edge row
    int s = edges[2 * e];
    int t = edges[2 * e + 1];

    // prefetch next tile's edge indices (global_prefetch_b8)
    long ntile = tile + stride;
    if (ntile < numTiles)
      __builtin_prefetch(edges + (ntile * 16 + lm) * 2, 0, 0);

    v8f macc[4], aacc[4];
#pragma unroll
    for (int nt = 0; nt < 4; ++nt) { macc[nt] = vzero8(); aacc[nt] = vzero8(); }

    // ---- layer 1: K = 160 (5 k-tiles), N = 64 (4 n-tiles), shared A ----
#pragma unroll
    for (int kt = 0; kt < 5; ++kt) {
      int k0 = kt * 32 + half * 8;
      v16bf a = make_frag(ei_ptr(nsb, efb, s, t, e, k0),
                          ei_ptr(nsb, efb, s, t, e, k0 + 16));
#pragma unroll
      for (int nt = 0; nt < 4; ++nt) {
        v16bf bm = load_bfrag(smw + (size_t)(kt * 4 + nt) * 512, lane);
        macc[nt] = __builtin_amdgcn_wmma_f32_16x16x32_bf16(
            false, a, false, bm, (short)0, macc[nt], false, false);
        v16bf ba = load_bfrag(smw + 10240 + (size_t)(kt * 4 + nt) * 512, lane);
        aacc[nt] = __builtin_amdgcn_wmma_f32_16x16x32_bf16(
            false, a, false, ba, (short)0, aacc[nt], false, false);
      }
    }

    // ---- stage msg hidden (bias + relu), reload as A fragments ----
#pragma unroll
    for (int nt = 0; nt < 4; ++nt) {
      float b = mb1[nt * 16 + lm];
#pragma unroll
      for (int r = 0; r < 8; ++r) {
        float v = fmaxf(macc[nt][r] + b, 0.f);
        hbase[(r + 8 * half) * HSTRIDE + nt * 16 + lm] = (bf16_t)v;
      }
    }
    v16bf a2m[2];
#pragma unroll
    for (int kt2 = 0; kt2 < 2; ++kt2) {
      const bf16_t* p = hbase + lm * HSTRIDE + kt2 * 32 + half * 8;
      a2m[kt2] = make_frag(p, p + 16);
    }

    // ---- stage att hidden (reuse LDS region) ----
#pragma unroll
    for (int nt = 0; nt < 4; ++nt) {
      float b = ab1[nt * 16 + lm];
#pragma unroll
      for (int r = 0; r < 8; ++r) {
        float v = fmaxf(aacc[nt][r] + b, 0.f);
        hbase[(r + 8 * half) * HSTRIDE + nt * 16 + lm] = (bf16_t)v;
      }
    }
    v16bf a2a[2];
#pragma unroll
    for (int kt2 = 0; kt2 < 2; ++kt2) {
      const bf16_t* p = hbase + lm * HSTRIDE + kt2 * 32 + half * 8;
      a2a[kt2] = make_frag(p, p + 16);
    }

    // ---- layer 2: K = 64 (2 k-tiles), N = 64 ----
    v8f dacc[4], gacc[4];
#pragma unroll
    for (int nt = 0; nt < 4; ++nt) { dacc[nt] = vzero8(); gacc[nt] = vzero8(); }
#pragma unroll
    for (int kt2 = 0; kt2 < 2; ++kt2) {
#pragma unroll
      for (int nt = 0; nt < 4; ++nt) {
        v16bf bm = load_bfrag(smw + 20480 + (size_t)(kt2 * 4 + nt) * 512, lane);
        dacc[nt] = __builtin_amdgcn_wmma_f32_16x16x32_bf16(
            false, a2m[kt2], false, bm, (short)0, dacc[nt], false, false);
        v16bf ba = load_bfrag(smw + 24576 + (size_t)(kt2 * 4 + nt) * 512, lane);
        gacc[nt] = __builtin_amdgcn_wmma_f32_16x16x32_bf16(
            false, a2a[kt2], false, ba, (short)0, gacc[nt], false, false);
      }
    }

    // ---- draft * sigmoid(gate), segment-sum scatter via f32 atomics ----
#pragma unroll
    for (int r = 0; r < 8; ++r) {
      int rr = r + 8 * half;               // C layout row of this element
      int trow = __shfl(t, rr, 32);        // tgt of edge row rr
      float* dst = msum + (long)trow * 64 + lm;
#pragma unroll
      for (int nt = 0; nt < 4; ++nt) {
        float d = dacc[nt][r] + mb2[nt * 16 + lm];
        float g = sigmoidf_(gacc[nt][r] + ab2[nt * 16 + lm]);
        atomicAdd(dst + nt * 16, d * g);
      }
    }
  }
}

// ---------------------------------------------------------------------------
// WMMA GRU kernel: per 16-node tile compute
//   gi = msum @ wih.T + bih ; gh = ns @ whh.T + bhh (K=64, N=192)
// Fused accumulation: (ir+hr), (iz+hz) share accumulators via WMMA C-chaining;
// ig / hg kept separate (r gates hg alone). 48 WMMAs / tile.
// ---------------------------------------------------------------------------
#define GRU_WAVES 8

__global__ void __launch_bounds__(256) gru_wmma_kernel(
    const bf16_t* __restrict__ msumb, const bf16_t* __restrict__ nsb,
    const float* __restrict__ nsf,
    const float* __restrict__ wih, const float* __restrict__ whh,
    const float* __restrict__ bih, const float* __restrict__ bhh,
    float* __restrict__ out, long numTiles)
{
  // wih.T frags [0:12288), whh.T frags [12288:24576)  (K=64 -> nkt=2, NT=12)
  __shared__ __align__(16) bf16_t smw[24576];
  const int tid = threadIdx.x;
  pack_b_frags(smw,         wih, 2, 12, true, 64, tid, 256);
  pack_b_frags(smw + 12288, whh, 2, 12, true, 64, tid, 256);
  __syncthreads();

  const int wave = tid >> 5;
  const int lane = tid & 31;
  const int half = lane >> 4;
  const int lm   = lane & 15;

  const long stride = (long)gridDim.x * GRU_WAVES;
  for (long tile = (long)blockIdx.x * GRU_WAVES + wave; tile < numTiles;
       tile += stride) {
    long row = tile * 16 + lm;     // this lane's node row for A fragments

    v8f rz[8], igv[4], hgv[4];
#pragma unroll
    for (int q = 0; q < 8; ++q) rz[q] = vzero8();
#pragma unroll
    for (int q = 0; q < 4; ++q) { igv[q] = vzero8(); hgv[q] = vzero8(); }

#pragma unroll
    for (int kt = 0; kt < 2; ++kt) {
      int k0 = kt * 32 + half * 8;
      const bf16_t* pm = msumb + row * 64 + k0;
      const bf16_t* ph = nsb   + row * 64 + k0;
      v16bf am = make_frag(pm, pm + 16);
      v16bf ah = make_frag(ph, ph + 16);
#pragma unroll
      for (int nt = 0; nt < 8; ++nt) {
        v16bf bi = load_bfrag(smw + (size_t)(kt * 12 + nt) * 512, lane);
        rz[nt] = __builtin_amdgcn_wmma_f32_16x16x32_bf16(
            false, am, false, bi, (short)0, rz[nt], false, false);
        v16bf bh = load_bfrag(smw + 12288 + (size_t)(kt * 12 + nt) * 512, lane);
        rz[nt] = __builtin_amdgcn_wmma_f32_16x16x32_bf16(
            false, ah, false, bh, (short)0, rz[nt], false, false);
      }
#pragma unroll
      for (int nt = 8; nt < 12; ++nt) {
        v16bf bi = load_bfrag(smw + (size_t)(kt * 12 + nt) * 512, lane);
        igv[nt - 8] = __builtin_amdgcn_wmma_f32_16x16x32_bf16(
            false, am, false, bi, (short)0, igv[nt - 8], false, false);
        v16bf bh = load_bfrag(smw + 12288 + (size_t)(kt * 12 + nt) * 512, lane);
        hgv[nt - 8] = __builtin_amdgcn_wmma_f32_16x16x32_bf16(
            false, ah, false, bh, (short)0, hgv[nt - 8], false, false);
      }
    }

    // gate math + state update (f32)
#pragma unroll
    for (int nt = 0; nt < 4; ++nt) {
      int j = nt * 16 + lm;                 // state column
      float br  = bih[j]       + bhh[j];
      float bz  = bih[64 + j]  + bhh[64 + j];
      float bim = bih[128 + j];
      float bhm = bhh[128 + j];
#pragma unroll
      for (int r = 0; r < 8; ++r) {
        int rr = r + 8 * half;
        long n = tile * 16 + rr;
        float rg = sigmoidf_(rz[nt][r] + br);
        float zg = sigmoidf_(rz[nt + 4][r] + bz);
        float ng = tanhf((igv[nt][r] + bim) + rg * (hgv[nt][r] + bhm));
        float h  = nsf[n * 64 + j];
        out[n * 64 + j] = (1.f - zg) * ng + zg * h;
      }
    }
  }
}

// ---------------------------------------------------------------------------
// Node-side scalar kernels (small vs. edge MLP / GRU)
// ---------------------------------------------------------------------------

// out = relu(in@w1+b1)@w2+b2 ; dims fixed 64 -> 128 -> 64 (input MLP & readout)
__global__ void __launch_bounds__(128) mlp2_kernel(
    const float* __restrict__ in,
    const float* __restrict__ w1, const float* __restrict__ b1,
    const float* __restrict__ w2, const float* __restrict__ b2,
    float* __restrict__ out)
{
  __shared__ float x[64];
  __shared__ float h[128];
  const long n = blockIdx.x;
  const int tid = threadIdx.x;
  if (tid < 64) x[tid] = in[n * 64 + tid];
  __syncthreads();
  float acc = b1[tid];
  for (int k = 0; k < 64; ++k) acc += x[k] * w1[k * 128 + tid];
  h[tid] = fmaxf(acc, 0.f);
  __syncthreads();
  if (tid < 64) {
    float o = b2[tid];
    for (int k = 0; k < 128; ++k) o += h[k] * w2[k * 64 + tid];
    out[n * 64 + tid] = o;
  }
}

// ns_new = concat([ns, old]) @ res_w + res_b   (N x 128 @ 128 x 64)
__global__ void __launch_bounds__(256) residual_kernel(
    const float* __restrict__ ns, const float* __restrict__ old,
    const float* __restrict__ rw, const float* __restrict__ rb,
    float* __restrict__ out)
{
  __shared__ float a[4][64];
  __shared__ float b[4][64];
  const int tid = threadIdx.x;
  const int ln = tid >> 6;
  const int j  = tid & 63;
  const long n = (long)blockIdx.x * 4 + ln;
  a[ln][j] = ns[n * 64 + j];
  b[ln][j] = old[n * 64 + j];
  __syncthreads();
  float o = rb[j];
  for (int k = 0; k < 64; ++k) {
    o += a[ln][k] * rw[k * 64 + j];
    o += b[ln][k] * rw[(64 + k) * 64 + j];
  }
  out[n * 64 + j] = o;
}

__global__ void convert_kernel(const float* __restrict__ in,
                               bf16_t* __restrict__ out, long n) {
  long i = (long)blockIdx.x * 256 + threadIdx.x;
  if (i < n) out[i] = (bf16_t)in[i];
}

__global__ void relu_kernel(float* __restrict__ p, long n) {
  long i = (long)blockIdx.x * 256 + threadIdx.x;
  if (i < n) p[i] = fmaxf(p[i], 0.f);
}

__global__ void zero_kernel(float* __restrict__ p, long n) {
  long i = (long)blockIdx.x * 256 + threadIdx.x;
  if (i < n) p[i] = 0.f;
}

// ---------------------------------------------------------------------------
// Host-side orchestration (graph-capture safe: kernels + D2D memcpyAsync only)
// ---------------------------------------------------------------------------
extern "C" void kernel_launch(void* const* d_in, const int* in_sizes, int n_in,
                              void* d_out, int out_size, void* d_ws, size_t ws_size,
                              hipStream_t stream)
{
  (void)in_sizes; (void)n_in; (void)out_size; (void)ws_size;
  const int N = 100000;
  const int E = 1600000;

  const float* X      = (const float*)d_in[0];
  const int*   edges  = (const int*)  d_in[1];
  const float* EF     = (const float*)d_in[2];
  const float* in_w1  = (const float*)d_in[3];
  const float* in_b1  = (const float*)d_in[4];
  const float* in_w2  = (const float*)d_in[5];
  const float* in_b2  = (const float*)d_in[6];
  const float* msg_w1 = (const float*)d_in[7];
  const float* msg_b1 = (const float*)d_in[8];
  const float* msg_w2 = (const float*)d_in[9];
  const float* msg_b2 = (const float*)d_in[10];
  const float* att_w1 = (const float*)d_in[11];
  const float* att_b1 = (const float*)d_in[12];
  const float* att_w2 = (const float*)d_in[13];
  const float* att_b2 = (const float*)d_in[14];
  const float* res_w  = (const float*)d_in[15];
  const float* res_b  = (const float*)d_in[16];
  const float* g_wih  = (const float*)d_in[17];
  const float* g_whh  = (const float*)d_in[18];
  const float* g_bih  = (const float*)d_in[19];
  const float* g_bhh  = (const float*)d_in[20];
  const float* ro_w1  = (const float*)d_in[21];
  const float* ro_b1  = (const float*)d_in[22];
  const float* ro_w2  = (const float*)d_in[23];
  const float* ro_b2  = (const float*)d_in[24];

  char* ws = (char*)d_ws;
  size_t o = 0;
  float*  ns_a  = (float*)(ws + o);  o += (size_t)N * 64 * 4;
  float*  ns_b  = (float*)(ws + o);  o += (size_t)N * 64 * 4;
  float*  oldb  = (float*)(ws + o);  o += (size_t)N * 64 * 4;
  float*  msum  = (float*)(ws + o);  o += (size_t)N * 64 * 4;
  bf16_t* nsb   = (bf16_t*)(ws + o); o += (size_t)N * 64 * 2;
  bf16_t* msumb = (bf16_t*)(ws + o); o += (size_t)N * 64 * 2;
  bf16_t* efb   = (bf16_t*)(ws + o); o += (size_t)E * 32 * 2;

  const long nsElems = (long)N * 64;
  const long efElems = (long)E * 32;
  const int nsBlocks = (int)((nsElems + 255) / 256);
  const int efBlocks = (int)((efElems + 255) / 256);
  const long edgeTiles = E / 16;      // 100000
  const long nodeTiles = N / 16;      // 6250
  const int edgeBlocks = 1250;        // persistent: ~10 tiles / wave
  const int gruBlocks  = 200;         // persistent: ~4 tiles / wave

  // edges_feature -> bf16 (once)
  convert_kernel<<<dim3(efBlocks), 256, 0, stream>>>(EF, efb, efElems);

  // input MLP
  mlp2_kernel<<<dim3(N), 128, 0, stream>>>(X, in_w1, in_b1, in_w2, in_b2, ns_a);

  float* cur = ns_a;
  float* alt = ns_b;
  for (int i = 0; i < 2; ++i) {
    hipMemcpyAsync(oldb, cur, (size_t)N * 64 * 4, hipMemcpyDeviceToDevice, stream);
    if (i > 0)
      relu_kernel<<<dim3(nsBlocks), 256, 0, stream>>>(cur, nsElems);
    for (int p = 0; p < 2; ++p) {
      convert_kernel<<<dim3(nsBlocks), 256, 0, stream>>>(cur, nsb, nsElems);
      zero_kernel<<<dim3(nsBlocks), 256, 0, stream>>>(msum, nsElems);
      edge_kernel<<<dim3(edgeBlocks), 256, 0, stream>>>(
          nsb, efb, edges,
          msg_w1 + (size_t)i * 160 * 64, msg_b1 + (size_t)i * 64,
          msg_w2 + (size_t)i * 64 * 64,  msg_b2 + (size_t)i * 64,
          att_w1 + (size_t)i * 160 * 64, att_b1 + (size_t)i * 64,
          att_w2 + (size_t)i * 64 * 64,  att_b2 + (size_t)i * 64,
          msum, edgeTiles);
      convert_kernel<<<dim3(nsBlocks), 256, 0, stream>>>(msum, msumb, nsElems);
      gru_wmma_kernel<<<dim3(gruBlocks), 256, 0, stream>>>(
          msumb, nsb, cur,
          g_wih + (size_t)i * 192 * 64, g_whh + (size_t)i * 192 * 64,
          g_bih + (size_t)i * 192,      g_bhh + (size_t)i * 192,
          alt, nodeTiles);
      { float* tmp = cur; cur = alt; alt = tmp; }
    }
    residual_kernel<<<dim3(N / 4), 256, 0, stream>>>(
        cur, oldb, res_w + (size_t)i * 128 * 64, res_b + (size_t)i * 64, alt);
    { float* tmp = cur; cur = alt; alt = tmp; }
  }

  // readout
  mlp2_kernel<<<dim3(N), 128, 0, stream>>>(cur, ro_w1, ro_b1, ro_w2, ro_b2,
                                           (float*)d_out);
}
